// WindowAttention_90417651516018
// MI455X (gfx1250) — compile-verified, matrix operand order checked by reading, not run on hardware
//
#include <hip/hip_runtime.h>

#define K2V    49
#define DIMC   512
#define NHEADS 16
#define SCALEV 0.17677669529663687f   // 32^-0.5

typedef __attribute__((ext_vector_type(16))) __bf16 v16bf;
typedef __attribute__((ext_vector_type(2)))  __bf16 v2bf;
typedef __attribute__((ext_vector_type(8)))  float  v8f;

union Frag {
  v16bf          v;
  unsigned short h[16];
  unsigned int   u[8];
  uint4          q[2];
};

#if defined(__has_builtin)
#if __has_builtin(__builtin_amdgcn_cvt_pk_bf16_f32)
#define HAVE_CVT_PK_BF16 1
#endif
#endif

// pack two f32 -> packed bf16x2 (one v_cvt_pk_bf16_f32 when available)
__device__ __forceinline__ unsigned pack2bf(float lo, float hi) {
#ifdef HAVE_CVT_PK_BF16
  union { v2bf v; unsigned u; } c;
  c.v = __builtin_amdgcn_cvt_pk_bf16_f32(lo, hi);
  return c.u;
#else
  union { __bf16 b[2]; unsigned u; } c;
  c.b[0] = (__bf16)lo;
  c.b[1] = (__bf16)hi;
  return c.u;
#endif
}
__device__ __forceinline__ unsigned short f2bf(float x) {
  union { __bf16 b; unsigned short u; } c;
  c.b = (__bf16)x;
  return c.u;
}
__device__ __forceinline__ float bf2f(unsigned short h) {
  return __uint_as_float(((unsigned int)h) << 16);
}
__device__ __forceinline__ v8f vzero() {
  v8f z;
#pragma unroll
  for (int i = 0; i < 8; ++i) z[i] = 0.0f;
  return z;
}
__device__ __forceinline__ v8f wmma_bf16(const Frag& a, const Frag& b, v8f c) {
  return __builtin_amdgcn_wmma_f32_16x16x32_bf16(false, a.v, false, b.v,
                                                 (short)0, c, false, false);
}

// 16x32 bf16 fragment from LDS (row-major). Lane l (l<16): row row0+l,
// K = k0+[0..7] and k0+16+[0..7]; lanes 16..31: K = k0+8.. and k0+24..
__device__ __forceinline__ Frag ldsFrag(const unsigned short* base, int row0,
                                        int stride, int k0, int lane) {
  Frag f;
  const unsigned short* p = base + (row0 + (lane & 15)) * stride + k0 + ((lane >> 4) << 3);
  f.q[0] = *(const uint4*)(p);
  f.q[1] = *(const uint4*)(p + 16);
  return f;
}

// bf16 fragment from pre-converted global weights (row-major, stride DIMC)
__device__ __forceinline__ Frag gblFragB16(const unsigned short* W, int n, int k0, int lane) {
  Frag f;
  const unsigned short* p = W + n * DIMC + k0 + ((lane >> 4) << 3);
  f.q[0] = *(const uint4*)(p);
  f.q[1] = *(const uint4*)(p + 16);
  return f;
}

// convert 8+8 contiguous f32 (chunks at p, p+16) into a bf16 fragment slice
__device__ __forceinline__ Frag cvtFrag(const float* p) {
  Frag f;
  float4 c0 = *(const float4*)(p);
  float4 c1 = *(const float4*)(p + 4);
  float4 c2 = *(const float4*)(p + 16);
  float4 c3 = *(const float4*)(p + 20);
  f.u[0] = pack2bf(c0.x, c0.y); f.u[1] = pack2bf(c0.z, c0.w);
  f.u[2] = pack2bf(c1.x, c1.y); f.u[3] = pack2bf(c1.z, c1.w);
  f.u[4] = pack2bf(c2.x, c2.y); f.u[5] = pack2bf(c2.z, c2.w);
  f.u[6] = pack2bf(c3.x, c3.y); f.u[7] = pack2bf(c3.z, c3.w);
  return f;
}

template <bool PRE>
__device__ __forceinline__ Frag loadB(const float* Wf, const unsigned short* Wb,
                                      int n, int k0, int lane) {
  if (PRE) return gblFragB16(Wb, n, k0, lane);
  return cvtFrag(Wf + n * DIMC + k0 + ((lane >> 4) << 3));
}

// C = X[64x512](LDS bf16) * W^T (+bias)(*scale for first scaleTiles col-tiles) -> bf16 LDS.
// transStore=false: dst[token*1024 + n]; transStore=true: dst[n*64 + token].
template <bool PRE>
__device__ __forceinline__ void gemm_x(const unsigned short* XS,
                                       const float* Wf, const unsigned short* Wb,
                                       const float* bias, unsigned short* dst,
                                       int ntiles, int scaleTiles, float scaleVal,
                                       bool transStore, int wave, int lane) {
  const int lcol = lane & 15;
  const int hi8  = (lane >> 4) << 3;
  const int tpw  = ntiles >> 3;
#pragma unroll 1
  for (int cp = 0; cp < tpw; cp += 2) {
    const int ct0 = wave * tpw + cp;
    const int n0  = ct0 * 16 + lcol;
    const int n1  = n0 + 16;
    v8f acc0[4], acc1[4];
#pragma unroll
    for (int rt = 0; rt < 4; ++rt) { acc0[rt] = vzero(); acc1[rt] = vzero(); }
#pragma unroll 1
    for (int kt = 0; kt < 16; ++kt) {
      const int k0 = kt * 32;
      Frag a0 = ldsFrag(XS, 0,  DIMC, k0, lane);
      Frag a1 = ldsFrag(XS, 16, DIMC, k0, lane);
      Frag a2 = ldsFrag(XS, 32, DIMC, k0, lane);
      Frag a3 = ldsFrag(XS, 48, DIMC, k0, lane);
      Frag b0 = loadB<PRE>(Wf, Wb, n0, k0, lane);
      acc0[0] = wmma_bf16(a0, b0, acc0[0]);
      acc0[1] = wmma_bf16(a1, b0, acc0[1]);
      acc0[2] = wmma_bf16(a2, b0, acc0[2]);
      acc0[3] = wmma_bf16(a3, b0, acc0[3]);
      Frag b1 = loadB<PRE>(Wf, Wb, n1, k0, lane);
      acc1[0] = wmma_bf16(a0, b1, acc1[0]);
      acc1[1] = wmma_bf16(a1, b1, acc1[1]);
      acc1[2] = wmma_bf16(a2, b1, acc1[2]);
      acc1[3] = wmma_bf16(a3, b1, acc1[3]);
    }
#pragma unroll
    for (int c = 0; c < 2; ++c) {
      v8f* acc = c ? acc1 : acc0;
      const int n  = c ? n1 : n0;
      const int ct = ct0 + c;
      const float bv = bias ? bias[n] : 0.0f;
      const float sc = (ct < scaleTiles) ? scaleVal : 1.0f;
#pragma unroll
      for (int rt = 0; rt < 4; ++rt) {
        const int m0 = rt * 16 + hi8;
        if (transStore) {
          unsigned int* d = (unsigned int*)(dst + n * 64 + m0);
#pragma unroll
          for (int r = 0; r < 8; r += 2)
            d[r >> 1] = pack2bf((acc[rt][r] + bv) * sc, (acc[rt][r + 1] + bv) * sc);
        } else {
#pragma unroll
          for (int r = 0; r < 8; ++r)
            dst[(m0 + r) * 1024 + n] = f2bf((acc[rt][r] + bv) * sc);
        }
      }
    }
  }
}

template <bool PRE>
__global__ __launch_bounds__(256) void wattn_kernel(
    const float* __restrict__ x,       const float* __restrict__ qk_w,
    const float* __restrict__ qk_b,    const float* __restrict__ rev_v_w,
    const float* __restrict__ rev_v_b, const float* __restrict__ wg_w,
    const float* __restrict__ proj_w,  const float* __restrict__ proj_b,
    const float* __restrict__ rpb,     float* __restrict__ out,
    const unsigned short* __restrict__ wsb) {
  // 320 KB LDS: QK/em [64x1024] | V^T [512x64] | attn^T scratch [8x64x64] | X [64x512]
  __shared__ __align__(16) unsigned short smem[163840];
  unsigned short* QK = smem;               // 65536 elems
  unsigned short* VT = smem + 65536;       // 32768 elems
  unsigned short* SS = smem + 98304;       // 32768 elems
  unsigned short* XS = smem + 131072;      // 32768 elems
  unsigned short* PW = smem + 65536;       // reuses VT+SS: [64x1024]

  const unsigned short* qk_wb   = PRE ? wsb            : nullptr;
  const unsigned short* rev_v_wb= PRE ? wsb + 524288   : nullptr;
  const unsigned short* wg_wb   = PRE ? wsb + 786432   : nullptr;
  const unsigned short* proj_wb = PRE ? wsb + 1310720  : nullptr;

  const int b     = blockIdx.x;
  const float* xb = x + (size_t)b * (K2V * DIMC);
  const int wave  = threadIdx.x >> 5;
  const int lane  = threadIdx.x & 31;
  const int lcol  = lane & 15;
  const int hi8   = (lane >> 4) << 3;

  __builtin_prefetch(qk_w + threadIdx.x * 64, 0, 1);   // global_prefetch

  // 0) stage x -> LDS bf16 once (rows >= 49 zeroed)
#pragma unroll 1
  for (int i = threadIdx.x; i < 64 * 128; i += 256) {
    const int t  = i >> 7;
    const int c4 = (i & 127) << 2;
    uint2 v2; v2.x = 0u; v2.y = 0u;
    if (t < K2V) {
      const float4 xv = *(const float4*)(xb + t * DIMC + c4);
      v2.x = pack2bf(xv.x, xv.y);
      v2.y = pack2bf(xv.z, xv.w);
    }
    *(uint2*)(XS + t * DIMC + c4) = v2;
  }
  __syncthreads();

  // 1) QK = x @ qk_w^T + qk_b   (q half scaled by 1/sqrt(E))
  gemm_x<PRE>(XS, qk_w, qk_wb, qk_b, QK, 64, 32, SCALEV, false, wave, lane);
  // 2) V^T = (x @ rev_v_w^T + rev_v_b)^T
  gemm_x<PRE>(XS, rev_v_w, rev_v_wb, rev_v_b, VT, 32, 0, 1.0f, true, wave, lane);
  __syncthreads();

  // 3) attention: 2 heads per wave; em overwrites this head's q columns
#pragma unroll 1
  for (int hh = 0; hh < 2; ++hh) {
    const int h  = wave + (hh << 3);
    const int qc = h * 32;
    const int kc = 512 + h * 32;
    Frag qa[4];
#pragma unroll
    for (int rt = 0; rt < 4; ++rt) qa[rt] = ldsFrag(QK, rt * 16, 1024, qc, lane);
    v8f s[4][4];
#pragma unroll
    for (int ct = 0; ct < 4; ++ct) {
      Frag kb = ldsFrag(QK, ct * 16, 1024, kc, lane);
#pragma unroll
      for (int rt = 0; rt < 4; ++rt) s[rt][ct] = wmma_bf16(qa[rt], kb, vzero());
    }
    // relative position bias + column mask
#pragma unroll
    for (int rt = 0; rt < 4; ++rt) {
#pragma unroll
      for (int ct = 0; ct < 4; ++ct) {
        const int j = ct * 16 + lcol;
#pragma unroll
        for (int r = 0; r < 8; ++r) {
          const int i = rt * 16 + hi8 + r;
          float val = s[rt][ct][r];
          if (i < K2V && j < K2V) {
            const int qy = i / 7, qx = i - qy * 7;
            const int ky = j / 7, kx = j - ky * 7;
            val += rpb[((qy - ky + 6) * 13 + (qx - kx + 6)) * NHEADS + h];
          }
          if (j >= K2V) val = -1e30f;
          s[rt][ct][r] = val;
        }
      }
    }
    // softmax over j (row i spans one 16-lane half x 4 col-tiles)
#pragma unroll
    for (int rt = 0; rt < 4; ++rt) {
#pragma unroll
      for (int r = 0; r < 8; ++r) {
        float m = fmaxf(fmaxf(s[rt][0][r], s[rt][1][r]),
                        fmaxf(s[rt][2][r], s[rt][3][r]));
#pragma unroll
        for (int d = 8; d > 0; d >>= 1) m = fmaxf(m, __shfl_xor(m, d, 32));
        float sum = 0.0f;
#pragma unroll
        for (int ct = 0; ct < 4; ++ct) {
          const float e = __expf(s[rt][ct][r] - m);
          s[rt][ct][r] = e;
          sum += e;
        }
#pragma unroll
        for (int d = 8; d > 0; d >>= 1) sum += __shfl_xor(sum, d, 32);
        const float inv = 1.0f / sum;
#pragma unroll
        for (int ct = 0; ct < 4; ++ct) s[rt][ct][r] *= inv;
      }
    }
    // store attn^T to per-wave scratch (padded source rows i>=49 zeroed)
    unsigned short* Sw = SS + wave * 4096;
#pragma unroll
    for (int ct = 0; ct < 4; ++ct) {
      const int j = ct * 16 + lcol;
#pragma unroll
      for (int rt = 0; rt < 4; ++rt) {
        const int ib = rt * 16 + hi8;
        unsigned int* d = (unsigned int*)(Sw + j * 64 + ib);
#pragma unroll
        for (int r = 0; r < 8; r += 2) {
          const float v0 = (ib + r     < K2V) ? s[rt][ct][r]     : 0.0f;
          const float v1 = (ib + r + 1 < K2V) ? s[rt][ct][r + 1] : 0.0f;
          d[r >> 1] = pack2bf(v0, v1);
        }
      }
    }
    // em = attn^T @ v_h  -> write into q columns of QK (em layout [token][h*32+e])
#pragma unroll
    for (int et = 0; et < 2; ++et) {
      v8f e0 = vzero(), e1 = vzero(), e2 = vzero(), e3 = vzero();
#pragma unroll
      for (int kcu = 0; kcu < 2; ++kcu) {
        Frag vb = ldsFrag(VT, h * 32 + et * 16, 64, kcu * 32, lane);
        Frag a0 = ldsFrag(Sw, 0,  64, kcu * 32, lane);
        Frag a1 = ldsFrag(Sw, 16, 64, kcu * 32, lane);
        Frag a2 = ldsFrag(Sw, 32, 64, kcu * 32, lane);
        Frag a3 = ldsFrag(Sw, 48, 64, kcu * 32, lane);
        e0 = wmma_bf16(a0, vb, e0);
        e1 = wmma_bf16(a1, vb, e1);
        e2 = wmma_bf16(a2, vb, e2);
        e3 = wmma_bf16(a3, vb, e3);
      }
      const int col = qc + et * 16 + lcol;
#pragma unroll
      for (int r = 0; r < 8; ++r) {
        QK[(0  + hi8 + r) * 1024 + col] = f2bf(e0[r]);
        QK[(16 + hi8 + r) * 1024 + col] = f2bf(e1[r]);
        QK[(32 + hi8 + r) * 1024 + col] = f2bf(e2[r]);
        QK[(48 + hi8 + r) * 1024 + col] = f2bf(e3[r]);
      }
    }
  }
  __syncthreads();

  // 4) PW = x @ wg_w^T  (per-token 32x32 mixing matrices) into VT+SS region
  gemm_x<PRE>(XS, wg_w, wg_wb, nullptr, PW, 64, 0, 1.0f, false, wave, lane);
  __syncthreads();

  // 5) per-token mixing: out1[t, h*32+f] = sum_e em[t,h,e] * pw[t,e,f]
  //    out1 packed into the (dead) k columns of QK: col 512 + h*32 + f
#pragma unroll 1
  for (int item = threadIdx.x; item < K2V * NHEADS; item += 256) {
    const int t = item >> 4;
    const int h = item & 15;
    float acc[32];
#pragma unroll
    for (int f = 0; f < 32; ++f) acc[f] = 0.0f;
    const unsigned short* emp = QK + t * 1024 + h * 32;
    const unsigned short* pwp = PW + t * 1024;
#pragma unroll 2
    for (int e = 0; e < 32; ++e) {
      const float ev = bf2f(emp[e]);
      const unsigned int* pr = (const unsigned int*)(pwp + e * 32);
#pragma unroll
      for (int f2 = 0; f2 < 16; ++f2) {
        const unsigned int p = pr[f2];
        acc[2 * f2]     += ev * bf2f((unsigned short)(p & 0xFFFFu));
        acc[2 * f2 + 1] += ev * bf2f((unsigned short)(p >> 16));
      }
    }
    unsigned int* d = (unsigned int*)(QK + t * 1024 + 512 + h * 32);
#pragma unroll
    for (int f = 0; f < 32; f += 2) d[f >> 1] = pack2bf(acc[f], acc[f + 1]);
  }
  __syncthreads();

  // 6) out = out1 @ proj_w^T + proj_b  -> global f32
  {
    float* ob = out + (size_t)b * (K2V * DIMC);
#pragma unroll 1
    for (int cp = 0; cp < 4; cp += 2) {
      const int ct0 = wave * 4 + cp;
      const int n0  = ct0 * 16 + lcol;
      const int n1  = n0 + 16;
      v8f acc0[4], acc1[4];
#pragma unroll
      for (int rt = 0; rt < 4; ++rt) { acc0[rt] = vzero(); acc1[rt] = vzero(); }
#pragma unroll 1
      for (int kt = 0; kt < 16; ++kt) {
        const int k0 = 512 + kt * 32;  // out1 lives in cols 512..1023
        Frag a0 = ldsFrag(QK, 0,  1024, k0, lane);
        Frag a1 = ldsFrag(QK, 16, 1024, k0, lane);
        Frag a2 = ldsFrag(QK, 32, 1024, k0, lane);
        Frag a3 = ldsFrag(QK, 48, 1024, k0, lane);
        Frag b0 = loadB<PRE>(proj_w, proj_wb, n0, kt * 32, lane);
        acc0[0] = wmma_bf16(a0, b0, acc0[0]);
        acc0[1] = wmma_bf16(a1, b0, acc0[1]);
        acc0[2] = wmma_bf16(a2, b0, acc0[2]);
        acc0[3] = wmma_bf16(a3, b0, acc0[3]);
        Frag b1 = loadB<PRE>(proj_w, proj_wb, n1, kt * 32, lane);
        acc1[0] = wmma_bf16(a0, b1, acc1[0]);
        acc1[1] = wmma_bf16(a1, b1, acc1[1]);
        acc1[2] = wmma_bf16(a2, b1, acc1[2]);
        acc1[3] = wmma_bf16(a3, b1, acc1[3]);
      }
#pragma unroll
      for (int c = 0; c < 2; ++c) {
        v8f* acc = c ? acc1 : acc0;
        const int n = c ? n1 : n0;
        const float bv = proj_b[n];
#pragma unroll
        for (int rt = 0; rt < 4; ++rt) {
#pragma unroll
          for (int r = 0; r < 8; ++r) {
            const int t = rt * 16 + hi8 + r;
            if (t < K2V) ob[t * DIMC + n] = acc[rt][r] + bv;
          }
        }
      }
    }
  }
}

// f32 -> bf16 weight pre-conversion (pairs)
__global__ __launch_bounds__(256) void cvt_w_kernel(const float* __restrict__ src,
                                                    unsigned short* __restrict__ dst,
                                                    int n2) {
  const int i = blockIdx.x * 256 + threadIdx.x;
  if (i < n2) {
    const float2 v = *(const float2*)(src + 2 * i);
    *(unsigned int*)(dst + 2 * i) = pack2bf(v.x, v.y);
  }
}

extern "C" void kernel_launch(void* const* d_in, const int* in_sizes, int n_in,
                              void* d_out, int out_size, void* d_ws, size_t ws_size,
                              hipStream_t stream) {
  const float* x       = (const float*)d_in[0];
  const float* qk_w    = (const float*)d_in[1];
  const float* qk_b    = (const float*)d_in[2];
  const float* rev_v_w = (const float*)d_in[3];
  const float* rev_v_b = (const float*)d_in[4];
  const float* wg_w    = (const float*)d_in[5];
  const float* proj_w  = (const float*)d_in[6];
  const float* proj_b  = (const float*)d_in[7];
  const float* rpb     = (const float*)d_in[8];
  float* out = (float*)d_out;

  const int BW = in_sizes[0] / (K2V * DIMC);   // 2048

  const size_t WS_NEED = (size_t)(524288 + 262144 + 524288 + 262144) * 2;  // 3 MB
  if (d_ws && ws_size >= WS_NEED) {
    unsigned short* wsb = (unsigned short*)d_ws;
    cvt_w_kernel<<<dim3(524288 / 2 / 256), dim3(256), 0, stream>>>(qk_w,    wsb,            524288 / 2);
    cvt_w_kernel<<<dim3(262144 / 2 / 256), dim3(256), 0, stream>>>(rev_v_w, wsb + 524288,   262144 / 2);
    cvt_w_kernel<<<dim3(524288 / 2 / 256), dim3(256), 0, stream>>>(wg_w,    wsb + 786432,   524288 / 2);
    cvt_w_kernel<<<dim3(262144 / 2 / 256), dim3(256), 0, stream>>>(proj_w,  wsb + 1310720,  262144 / 2);
    wattn_kernel<true><<<dim3(BW), dim3(256), 0, stream>>>(
        x, qk_w, qk_b, rev_v_w, rev_v_b, wg_w, proj_w, proj_b, rpb, out, wsb);
  } else {
    wattn_kernel<false><<<dim3(BW), dim3(256), 0, stream>>>(
        x, qk_w, qk_b, rev_v_w, rev_v_b, wg_w, proj_w, proj_b, rpb, out, nullptr);
  }
}
